// DSGEA_19971597926542
// MI455X (gfx1250) — matrix operand order repaired and checked by reading, not compiled, not run on hardware
//
#include <hip/hip_runtime.h>

// ---------------- types for CDNA5 WMMA ----------------
typedef __bf16 bf16_t;
typedef __attribute__((ext_vector_type(16))) __bf16        v16bf;
typedef __attribute__((ext_vector_type(8)))  float         v8f;
typedef __attribute__((ext_vector_type(4)))  unsigned int  u32x4;

#define DEV static __device__ __forceinline__

DEV float leakyf(float x) { return x > 0.f ? x : 0.01f * x; }

DEV float warp_sum(float v) {
  #pragma unroll
  for (int m = 16; m; m >>= 1) v += __shfl_xor(v, m, 32);
  return v;
}

DEV float atomicMaxF(float* addr, float val) {
  int* ia = (int*)addr;
  int old = __float_as_int(*addr);
  while (val > __int_as_float(old)) {
    int assumed = old;
    old = atomicCAS(ia, assumed, __float_as_int(val));
    if (old == assumed) break;
  }
  return __int_as_float(old);
}

// ---------------- WMMA GEMM: C[M,Nout] = A[M,K] * B[Nout,K]^T ----------------
// A row-major bf16, lda = K (mult of 32), rows zero-padded to mult of 16.
// B row-major bf16 (weight rows), ldb = K, rows zero-padded to mult of 64.
// One wave computes a 16x64 strip: 1 A fragment reused across 4 B fragments,
// 4x v_wmma_f32_16x16x32_bf16 per K-step, no EXEC masking anywhere near WMMA.
__global__ void k_wmma_gemm4(const bf16_t* __restrict__ A, int lda,
                             const bf16_t* __restrict__ B, int ldb,
                             float* __restrict__ C, int ldc,
                             int M, int Nout, int K, int act) {
  const int lane   = threadIdx.x & 31;
  const int wvid   = threadIdx.x >> 5;
  const int mtile  = blockIdx.x;
  const int ngrp   = blockIdx.y * (blockDim.x >> 5) + wvid;   // group of 4 n-tiles
  const int ntile0 = ngrp * 4;
  const int ntiles = (Nout + 15) >> 4;
  if (ntile0 >= ntiles) return;        // wave-uniform exit

  const int l15  = lane & 15;
  const int hi   = lane >> 4;          // 0 or 1
  // A(16x32 bf16) layout: lanes0-15 row M=l15, K {0..7,16..23}; lanes16-31 same rows +8
  const int aoff = hi ? 8 : 0;
  // B(32x16) layout: lanes0-15 col N=l15, K {0..7,16..23}; lanes16-31 col N=l15, K {8..15,24..31}
  const int boff = hi ? 8 : 0;

  const bf16_t* arow = A + (size_t)(mtile * 16 + l15) * lda;   // padded: always valid
  const bf16_t* brow[4];
  #pragma unroll
  for (int t = 0; t < 4; ++t)
    brow[t] = B + (size_t)((ntile0 + t) * 16 + l15) * ldb;     // padded: always valid

  union Frag { v16bf v; u32x4 q[2]; };
  const v8f vzero = {};
  v8f acc[4];
  #pragma unroll
  for (int t = 0; t < 4; ++t) acc[t] = vzero;

  for (int k0 = 0; k0 < K; k0 += 32) {
    Frag a;
    a.q[0] = *(const u32x4*)(arow + k0 + aoff);
    a.q[1] = *(const u32x4*)(arow + k0 + 16 + aoff);
    #pragma unroll
    for (int t = 0; t < 4; ++t) {
      Frag b;
      b.q[0] = *(const u32x4*)(brow[t] + k0 + boff);
      b.q[1] = *(const u32x4*)(brow[t] + k0 + 16 + boff);
      acc[t] = __builtin_amdgcn_wmma_f32_16x16x32_bf16(false, a.v, false, b.v,
                                                       (short)0, acc[t], false, false);
    }
  }

  const int mbase = mtile * 16 + (hi ? 8 : 0);   // C/D layout: lanes16-31 hold M=8..15
  #pragma unroll
  for (int t = 0; t < 4; ++t) {
    int col = (ntile0 + t) * 16 + l15;
    if (col < Nout) {
      #pragma unroll
      for (int r = 0; r < 8; ++r) {
        int m = mbase + r;
        if (m < M) {
          float val = acc[t][r];
          if (act) val = fmaxf(val, 0.f);
          C[(size_t)m * ldc + col] = val;
        }
      }
    }
  }
}

// ---------------- small helper kernels ----------------
// zero-pads both K (ldd >= cols) and rows (rows_pad >= rows)
__global__ void k_to_bf16_pad(const float* __restrict__ src, bf16_t* __restrict__ dst,
                              int rows, int cols, int ldd, int rows_pad) {
  int i = blockIdx.x * blockDim.x + threadIdx.x;
  int total = rows_pad * ldd;
  if (i >= total) return;
  int r = i / ldd, c = i - r * ldd;
  dst[i] = (r < rows && c < cols) ? (bf16_t)src[(size_t)r * cols + c] : (bf16_t)0.f;
}

__global__ void k_fill(float* p, float v, int n) {
  int i = blockIdx.x * blockDim.x + threadIdx.x;
  if (i < n) p[i] = v;
}

__global__ void k_relu(float* p, int n) {
  int i = blockIdx.x * blockDim.x + threadIdx.x;
  if (i < n) p[i] = fmaxf(p[i], 0.f);
}

__global__ void k_relu_region(float* p, int rows, int cols, int ld, int coff) {
  int i = blockIdx.x * blockDim.x + threadIdx.x;
  if (i >= rows * cols) return;
  int r = i / cols, c = i - r * cols;
  float* q = p + (size_t)r * ld + coff + c;
  *q = fmaxf(*q, 0.f);
}

__global__ void k_copy_cols(const float* __restrict__ src, int lds,
                            float* __restrict__ dst, int ldd,
                            int rows, int cols, int dcol) {
  int i = blockIdx.x * blockDim.x + threadIdx.x;
  if (i >= rows * cols) return;
  int r = i / cols, c = i - r * cols;
  dst[(size_t)r * ldd + dcol + c] = src[(size_t)r * lds + c];
}

__global__ void k_l2norm_rows(const float* __restrict__ src, float* __restrict__ dst,
                              int rows, int cols) {
  int w = (blockIdx.x * blockDim.x + threadIdx.x) >> 5;
  int lane = threadIdx.x & 31;
  if (w >= rows) return;
  const float* s = src + (size_t)w * cols;
  float ss = 0.f;
  for (int c = lane; c < cols; c += 32) { float t = s[c]; ss += t * t; }
  ss = warp_sum(ss);
  float inv = 1.f / fmaxf(sqrtf(ss), 1e-12f);
  float* d = dst + (size_t)w * cols;
  for (int c = lane; c < cols; c += 32) d[c] = s[c] * inv;
}

__global__ void k_deg(const int* __restrict__ idx, float* __restrict__ deg, int n) {
  int i = blockIdx.x * blockDim.x + threadIdx.x;
  if (i < n) atomicAdd(&deg[idx[i]], 1.f);
}

__global__ void k_dis(float* deg, int n) {
  int i = blockIdx.x * blockDim.x + threadIdx.x;
  if (i < n) { float d = deg[i]; deg[i] = d > 0.f ? rsqrtf(d) : 0.f; }
}

__global__ void k_gcn_scatter(const float* __restrict__ x,
                              const int* __restrict__ ej, const int* __restrict__ ei,
                              const float* __restrict__ dis,
                              float* __restrict__ out, int E, int D) {
  int e = (blockIdx.x * blockDim.x + threadIdx.x) >> 5;
  int lane = threadIdx.x & 31;
  if (e >= E) return;
  int j = ej[e], i = ei[e];
  float nrm = dis[j] * dis[i];
  const float* src = x + (size_t)j * D;
  float* dst = out + (size_t)i * D;
  for (int d = lane; d < D; d += 32) atomicAdd(&dst[d], nrm * src[d]);
}

__global__ void k_highway(const float* __restrict__ x1, const float* __restrict__ x2,
                          const float* __restrict__ logits, const float* __restrict__ b,
                          float* __restrict__ out, int rows, int cols) {
  int i = blockIdx.x * blockDim.x + threadIdx.x;
  if (i >= rows * cols) return;
  int c = i % cols;
  float g = 1.f / (1.f + __expf(-(logits[i] + b[c])));
  out[i] = g * x2[i] + (1.f - g) * x1[i];
}

__global__ void k_matvec(const float* __restrict__ X, const float* __restrict__ v,
                         float* __restrict__ out, int rows, int cols, int ld) {
  int w = (blockIdx.x * blockDim.x + threadIdx.x) >> 5;
  int lane = threadIdx.x & 31;
  if (w >= rows) return;
  const float* row = X + (size_t)w * ld;
  float acc = 0.f;
  for (int c = lane; c < cols; c += 32) acc += row[c] * v[c];
  acc = warp_sum(acc);
  if (lane == 0) out[w] = acc;
}

// segment softmax pieces
__global__ void k_segmax(const float* __restrict__ z, const int* __restrict__ idx,
                         float* __restrict__ m, int n) {
  int i = blockIdx.x * blockDim.x + threadIdx.x;
  if (i < n) atomicMaxF(&m[idx[i]], z[i]);
}
__global__ void k_segexp(const float* __restrict__ z, const int* __restrict__ idx,
                         const float* __restrict__ m, float* __restrict__ ex,
                         float* __restrict__ s, int n) {
  int i = blockIdx.x * blockDim.x + threadIdx.x;
  if (i >= n) return;
  float e = __expf(z[i] - m[idx[i]]);
  ex[i] = e;
  atomicAdd(&s[idx[i]], e);
}
__global__ void k_segnorm(float* __restrict__ ex, const int* __restrict__ idx,
                          const float* __restrict__ s, int n) {
  int i = blockIdx.x * blockDim.x + threadIdx.x;
  if (i < n) ex[i] = ex[i] / s[idx[i]];
}

// GAT-E edge kernels
__global__ void k_edge_e1(const int* __restrict__ im, const int* __restrict__ io,
                          const int* __restrict__ rel,
                          const float* __restrict__ sm, const float* __restrict__ so,
                          const float* __restrict__ rr, float* __restrict__ z, int E) {
  int i = blockIdx.x * blockDim.x + threadIdx.x;
  if (i >= E) return;
  float e = 0.5f * (sm[im[i]] + so[io[i]]) + rr[rel[i]];
  z[i] = leakyf(e);
}

__global__ void k_xedge(const float* __restrict__ xm, const int* __restrict__ im,
                        const float* __restrict__ alpha, float* __restrict__ xedge,
                        int E, int D) {
  int e = (blockIdx.x * blockDim.x + threadIdx.x) >> 5;
  int lane = threadIdx.x & 31;
  if (e >= E) return;
  const float* src = xm + (size_t)im[e] * D;
  float a = alpha[e];
  float vals[8];
  float ss = 0.f;
  int cnt = 0;
  for (int d = lane; d < D; d += 32) { float t = src[d] * a; vals[cnt++] = t; ss += t * t; }
  ss = warp_sum(ss);
  float inv = 1.f / fmaxf(sqrtf(ss), 1e-12f);
  cnt = 0;
  for (int d = lane; d < D; d += 32) xedge[(size_t)e * D + d] = vals[cnt++] * inv;
}

__global__ void k_edge_e2(const float* __restrict__ xedge,
                          const int* __restrict__ io, const int* __restrict__ rel,
                          const float* __restrict__ v2m, const float* __restrict__ v3m,
                          const float* __restrict__ s2o, const float* __restrict__ t3o,
                          const float* __restrict__ rv1, float* __restrict__ z,
                          int E, int D) {
  int e = (blockIdx.x * blockDim.x + threadIdx.x) >> 5;
  int lane = threadIdx.x & 31;
  if (e >= E) return;
  const float* xe = xedge + (size_t)e * D;
  float d2 = 0.f, d3 = 0.f;
  for (int d = lane; d < D; d += 32) { float t = xe[d]; d2 += t * v2m[d]; d3 += t * v3m[d]; }
  d2 = warp_sum(d2);
  d3 = warp_sum(d3);
  if (lane == 0) {
    int o = io[e];
    float er = 0.5f * (rv1[rel[e]] + 0.5f * (d3 + t3o[o]));
    z[e] = leakyf(s2o[o] + er + d2);
  }
}

__global__ void k_xclass_scatter(const float* __restrict__ xedge, const int* __restrict__ im,
                                 const int* __restrict__ cidx, const float* __restrict__ beta,
                                 float* __restrict__ xcls, int E, int D) {
  int e = (blockIdx.x * blockDim.x + threadIdx.x) >> 5;
  int lane = threadIdx.x & 31;
  if (e >= E) return;
  int c = cidx[e];
  float b = beta[e];
  const float* src = xedge + (size_t)im[e] * D;   // note: reference indexes x_edge by node id
  float* dst = xcls + (size_t)c * D;
  for (int d = lane; d < D; d += 32) atomicAdd(&dst[d], b * src[d]);
}

__global__ void k_ec(const float* __restrict__ xcls, const float* __restrict__ acv,
                     const float* __restrict__ t4, const int* __restrict__ cls_ent,
                     float* __restrict__ z, int C, int D) {
  int c = (blockIdx.x * blockDim.x + threadIdx.x) >> 5;
  int lane = threadIdx.x & 31;
  if (c >= C) return;
  const float* row = xcls + (size_t)c * D;
  float acc = 0.f;
  for (int d = lane; d < D; d += 32) acc += row[d] * acv[d];
  acc = warp_sum(acc);
  if (lane == 0) z[c] = leakyf(acc + t4[cls_ent[c]]);
}

__global__ void k_out_scatter(const float* __restrict__ xcls, const int* __restrict__ cls_ent,
                              const float* __restrict__ gama, float* __restrict__ out,
                              int C, int D) {
  int c = (blockIdx.x * blockDim.x + threadIdx.x) >> 5;
  int lane = threadIdx.x & 31;
  if (c >= C) return;
  int n = cls_ent[c];
  float g = gama[c];
  const float* src = xcls + (size_t)c * D;
  float* dst = out + (size_t)n * D;
  for (int d = lane; d < D; d += 32) atomicAdd(&dst[d], g * src[d]);
}

// final GAT
__global__ void k_gat_edge(const int* __restrict__ ei, const int* __restrict__ ej,
                           const float* __restrict__ si, const float* __restrict__ sj,
                           float* __restrict__ z, int E) {
  int e = blockIdx.x * blockDim.x + threadIdx.x;
  if (e < E) z[e] = leakyf(si[ei[e]] + sj[ej[e]]);
}

__global__ void k_gat_scatter(const float* __restrict__ x1, const int* __restrict__ ej,
                              const int* __restrict__ ei, const float* __restrict__ alpha,
                              float* __restrict__ out, int E, int D, int ldo, int coff) {
  int e = (blockIdx.x * blockDim.x + threadIdx.x) >> 5;
  int lane = threadIdx.x & 31;
  if (e >= E) return;
  float a = alpha[e];
  const float* src = x1 + (size_t)ej[e] * D;
  float* dst = out + (size_t)ei[e] * ldo + coff;
  for (int d = lane; d < D; d += 32) atomicAdd(&dst[d], a * src[d]);
}

// ---------------- host side ----------------
static inline int cdiv(long long a, long long b) { return (int)((a + b - 1) / b); }

extern "C" void kernel_launch(void* const* d_in, const int* in_sizes, int n_in,
                              void* d_out, int out_size, void* d_ws, size_t ws_size,
                              hipStream_t stream) {
  const int EH = 300, RH = 100, CH = 150;
  const int Nn = in_sizes[0] / EH;
  const int E  = in_sizes[2];
  const int EA = in_sizes[3] / 2;
  const int Cc = in_sizes[6];
  const int Rr = in_sizes[9] / RH;
  const int KpE = 320;                   // EH padded to mult of 32
  const int KpC = 160;                   // CH padded to mult of 32
  const int Mp  = cdiv(Nn, 16) * 16;     // A rows padded to tile
  const int NpE = cdiv(EH, 64) * 64;     // B rows padded to 4-tile group (320)
  const int NpC = cdiv(CH, 64) * 64;     // 192

  const float* x_e   = (const float*)d_in[0];
  const int*   eidx  = (const int*)d_in[1];
  const int*   rel   = (const int*)d_in[2];
  const int*   eall  = (const int*)d_in[3];
  const int*   trip  = (const int*)d_in[4];
  const int*   cih   = (const int*)d_in[5];
  const int*   hcls  = (const int*)d_in[6];
  const int*   cit   = (const int*)d_in[7];
  const int*   tcls  = (const int*)d_in[8];
  const float* remb  = (const float*)d_in[9];
  const float* hw1w  = (const float*)d_in[10];
  const float* hw1b  = (const float*)d_in[11];
  const float* hw2w  = (const float*)d_in[12];
  const float* hw2b  = (const float*)d_in[13];
  const float* h_ac  = (const float*)d_in[14];
  const float* h_ar  = (const float*)d_in[15];
  const float* h_wh  = (const float*)d_in[16];
  const float* h_wt  = (const float*)d_in[17];
  const float* h_hww = (const float*)d_in[18];
  const float* h_hwb = (const float*)d_in[19];
  const float* t_ac  = (const float*)d_in[20];
  const float* t_ar  = (const float*)d_in[21];
  const float* t_wh  = (const float*)d_in[22];
  const float* t_wt  = (const float*)d_in[23];
  const float* t_hww = (const float*)d_in[24];
  const float* t_hwb = (const float*)d_in[25];
  const float* gai   = (const float*)d_in[26];
  const float* gaj   = (const float*)d_in[27];

  const int* eh  = eidx;          // edge_index[0]
  const int* et  = eidx + E;      // edge_index[1]
  const int* jA  = eall;          // edge_index_all[0]
  const int* iA  = eall + EA;     // edge_index_all[1]

  // workspace bump allocator (256B aligned)
  size_t off = 0;
  auto alloc = [&](size_t bytes) -> void* {
    void* p = (char*)d_ws + off;
    off += (bytes + 255) & ~(size_t)255;
    return p;
  };
  float* f_x    = (float*)alloc((size_t)Nn * EH * 4);
  float* f_t1   = (float*)alloc((size_t)Nn * EH * 4);
  float* f_t2   = (float*)alloc((size_t)Nn * EH * 4);
  float* f_deg  = (float*)alloc((size_t)Nn * 4);
  float* f_remb = (float*)alloc((size_t)Rr * RH * 4);
  float* f_xh   = (float*)alloc((size_t)Nn * CH * 4);
  float* f_xt   = (float*)alloc((size_t)Nn * CH * 4);
  float* f_xeh  = (float*)alloc((size_t)Nn * CH * 4);
  float* f_xet  = (float*)alloc((size_t)Nn * CH * 4);
  float* f_s1   = (float*)alloc((size_t)Nn * 4);
  float* f_s2   = (float*)alloc((size_t)Nn * 4);
  float* f_s3   = (float*)alloc((size_t)Nn * 4);
  float* f_s4   = (float*)alloc((size_t)Nn * 4);
  float* f_r1   = (float*)alloc((size_t)Rr * 4);
  float* f_ez   = (float*)alloc((size_t)EA * 4);
  float* f_ea   = (float*)alloc((size_t)EA * 4);
  float* f_segm = (float*)alloc((size_t)E * 4);
  float* f_segs = (float*)alloc((size_t)E * 4);
  float* f_xedg = (float*)alloc((size_t)E * CH * 4);
  float* f_xcls = (float*)alloc((size_t)Cc * CH * 4);
  float* f_x1   = (float*)alloc((size_t)Nn * (EH + 2 * CH) * 4);   // N x 600
  bf16_t* bA    = (bf16_t*)alloc((size_t)Mp * KpE * 2);
  bf16_t* bA2   = (bf16_t*)alloc((size_t)Mp * KpC * 2);
  bf16_t* bW    = (bf16_t*)alloc((size_t)NpE * KpE * 2);
  float* out    = (float*)d_out;
  const int D600 = EH + 2 * CH;

  auto ew = [&](int n) { return dim3(cdiv(n, 256)); };                  // elementwise grid
  auto wv = [&](int rows) { return dim3(cdiv((long long)rows * 32, 256)); };  // wave/row grid

  auto conv = [&](const float* src, bf16_t* dst, int rows, int cols, int ldd, int rows_pad) {
    k_to_bf16_pad<<<ew(rows_pad * ldd), 256, 0, stream>>>(src, dst, rows, cols, ldd, rows_pad);
  };
  auto gemm = [&](const bf16_t* A, int lda, const bf16_t* B, int ldb,
                  float* C, int ldc, int M, int Nout, int K, int act) {
    int ngroups = cdiv(cdiv(Nout, 16), 4);              // 4 n-tiles per wave
    k_wmma_gemm4<<<dim3(cdiv(M, 16), cdiv(ngroups, 4)), 128, 0, stream>>>(
        A, lda, B, ldb, C, ldc, M, Nout, K, act);
  };
  auto seg_softmax = [&](const float* z, const int* idx, int n, int nseg, float* alpha) {
    k_fill<<<ew(nseg), 256, 0, stream>>>(f_segm, -3.0e38f, nseg);
    (void)hipMemsetAsync(f_segs, 0, (size_t)nseg * 4, stream);
    k_segmax<<<ew(n), 256, 0, stream>>>(z, idx, f_segm, n);
    k_segexp<<<ew(n), 256, 0, stream>>>(z, idx, f_segm, alpha, f_segs, n);
    k_segnorm<<<ew(n), 256, 0, stream>>>(alpha, idx, f_segs, n);
  };

  // 1. r_emb = l2norm(r_emb_table)
  k_l2norm_rows<<<wv(Rr), 256, 0, stream>>>(remb, f_remb, Rr, RH);

  // 2. degrees for GCN (edge_index_all, destination = row 1)
  (void)hipMemsetAsync(f_deg, 0, (size_t)Nn * 4, stream);
  k_deg<<<ew(EA), 256, 0, stream>>>(iA, f_deg, EA);
  k_dis<<<ew(Nn), 256, 0, stream>>>(f_deg, Nn);

  // 3. GCN layer 1 + highway 1
  (void)hipMemsetAsync(f_t1, 0, (size_t)Nn * EH * 4, stream);
  k_gcn_scatter<<<wv(EA), 256, 0, stream>>>(x_e, jA, iA, f_deg, f_t1, EA, EH);
  k_relu<<<ew(Nn * EH), 256, 0, stream>>>(f_t1, Nn * EH);
  conv(x_e, bA, Nn, EH, KpE, Mp);
  conv(hw1w, bW, EH, EH, KpE, NpE);
  gemm(bA, KpE, bW, KpE, f_t2, EH, Nn, EH, KpE, 0);
  k_highway<<<ew(Nn * EH), 256, 0, stream>>>(x_e, f_t1, f_t2, hw1b, f_x, Nn, EH);

  // 4. GCN layer 2 + highway 2
  (void)hipMemsetAsync(f_t1, 0, (size_t)Nn * EH * 4, stream);
  k_gcn_scatter<<<wv(EA), 256, 0, stream>>>(f_x, jA, iA, f_deg, f_t1, EA, EH);
  k_relu<<<ew(Nn * EH), 256, 0, stream>>>(f_t1, Nn * EH);
  conv(f_x, bA, Nn, EH, KpE, Mp);
  conv(hw2w, bW, EH, EH, KpE, NpE);
  gemm(bA, KpE, bW, KpE, f_t2, EH, Nn, EH, KpE, 0);
  k_highway<<<ew(Nn * EH), 256, 0, stream>>>(f_x, f_t1, f_t2, hw2b, f_x, Nn, EH);

  // bf16 copy of final x for the wh/wt projections of both GAT-E passes
  conv(f_x, bA, Nn, EH, KpE, Mp);

  // ---------- GAT-E pass ----------
  auto run_gat_e = [&](const int* im, const int* io, const int* cidx, const int* cls_ent,
                       const float* ac, const float* ar, const float* wwh, const float* wwt,
                       const float* hww, const float* hwb, bool main_is_head, float* dst) {
    // x_h = relu(x@wh.T), x_t = relu(x@wt.T)  (WMMA)
    conv(wwh, bW, CH, EH, KpE, NpC);
    gemm(bA, KpE, bW, KpE, f_xh, CH, Nn, CH, KpE, 1);
    conv(wwt, bW, CH, EH, KpE, NpC);
    gemm(bA, KpE, bW, KpE, f_xt, CH, Nn, CH, KpE, 1);

    const float* x_m = main_is_head ? f_xh : f_xt;
    const float* x_o = main_is_head ? f_xt : f_xh;
    const float* v1m = main_is_head ? ac + 0 * CH : ac + 4 * CH;
    const float* v2m = main_is_head ? ac + 1 * CH : ac + 5 * CH;
    const float* v3m = main_is_head ? ac + 2 * CH : ac + 6 * CH;
    const float* v1o = main_is_head ? ac + 4 * CH : ac + 0 * CH;
    const float* v2o = main_is_head ? ac + 5 * CH : ac + 1 * CH;
    const float* v3o = main_is_head ? ac + 6 * CH : ac + 2 * CH;
    const float* v4  = ac + 3 * CH;
    const float* acv = ac + 7 * CH;

    // e1 / alpha
    k_matvec<<<wv(Nn), 256, 0, stream>>>(x_m, v1m, f_s1, Nn, CH, CH);
    k_matvec<<<wv(Nn), 256, 0, stream>>>(x_o, v1o, f_s2, Nn, CH, CH);
    k_matvec<<<wv(Rr), 256, 0, stream>>>(f_remb, ar, f_r1, Rr, RH, RH);
    k_edge_e1<<<ew(E), 256, 0, stream>>>(im, io, rel, f_s1, f_s2, f_r1, f_ez, E);
    seg_softmax(f_ez, trip, E, E, f_ea);

    // x_edge = l2norm(x_m[im] * alpha)
    k_xedge<<<wv(E), 256, 0, stream>>>(x_m, im, f_ea, f_xedg, E, CH);

    // e / beta
    k_matvec<<<wv(Nn), 256, 0, stream>>>(x_o, v3o, f_s3, Nn, CH, CH);
    k_matvec<<<wv(Nn), 256, 0, stream>>>(x_o, v2o, f_s4, Nn, CH, CH);
    k_matvec<<<wv(Rr), 256, 0, stream>>>(f_remb, ar + RH, f_r1, Rr, RH, RH);
    k_edge_e2<<<wv(E), 256, 0, stream>>>(f_xedg, io, rel, v2m, v3m, f_s4, f_s3, f_r1,
                                         f_ez, E, CH);
    seg_softmax(f_ez, cidx, E, Cc, f_ea);

    // x_class
    (void)hipMemsetAsync(f_xcls, 0, (size_t)Cc * CH * 4, stream);
    k_xclass_scatter<<<wv(E), 256, 0, stream>>>(f_xedg, im, cidx, f_ea, f_xcls, E, CH);

    // e_c / gama
    k_matvec<<<wv(Nn), 256, 0, stream>>>(x_o, v4, f_s1, Nn, CH, CH);
    k_ec<<<wv(Cc), 256, 0, stream>>>(f_xcls, acv, f_s1, cls_ent, f_ez, Cc, CH);
    seg_softmax(f_ez, cls_ent, Cc, Nn, f_ea);

    // out = segment_sum(gama * x_class)
    (void)hipMemsetAsync(f_t1, 0, (size_t)Nn * CH * 4, stream);
    k_out_scatter<<<wv(Cc), 256, 0, stream>>>(f_xcls, cls_ent, f_ea, f_t1, Cc, CH);

    // highway(x_o, out)  (WMMA for the gate GEMM)
    conv(x_o, bA2, Nn, CH, KpC, Mp);
    conv(hww, bW, CH, CH, KpC, NpC);
    gemm(bA2, KpC, bW, KpC, f_t2, CH, Nn, CH, KpC, 0);
    k_highway<<<ew(Nn * CH), 256, 0, stream>>>(x_o, f_t1, f_t2, hwb, dst, Nn, CH);
  };

  // x_e_h: im=et, io=eh, main_is_head=false ; x_e_t: im=eh, io=et, main_is_head=true
  run_gat_e(et, eh, cih, hcls, h_ac, h_ar, h_wh, h_wt, h_hww, h_hwb, false, f_xeh);
  run_gat_e(eh, et, cit, tcls, t_ac, t_ar, t_wh, t_wt, t_hww, t_hwb, true,  f_xet);

  // x1 = concat([x, x_e_h, x_e_t])  (N x 600)
  k_copy_cols<<<ew(Nn * EH), 256, 0, stream>>>(f_x, EH, f_x1, D600, Nn, EH, 0);
  k_copy_cols<<<ew(Nn * CH), 256, 0, stream>>>(f_xeh, CH, f_x1, D600, Nn, CH, EH);
  k_copy_cols<<<ew(Nn * CH), 256, 0, stream>>>(f_xet, CH, f_x1, D600, Nn, CH, EH + CH);

  // final GAT over edge_index_all, output into d_out cols [300, 900)
  k_matvec<<<wv(Nn), 256, 0, stream>>>(f_x1, gai, f_s1, Nn, D600, D600);
  k_matvec<<<wv(Nn), 256, 0, stream>>>(f_x1, gaj, f_s2, Nn, D600, D600);
  k_gat_edge<<<ew(EA), 256, 0, stream>>>(iA, jA, f_s1, f_s2, f_ez, EA);
  seg_softmax(f_ez, iA, EA, Nn, f_ea);

  (void)hipMemsetAsync(out, 0, (size_t)Nn * (EH + D600) * 4, stream);
  k_copy_cols<<<ew(Nn * EH), 256, 0, stream>>>(f_x, EH, out, EH + D600, Nn, EH, 0);
  k_gat_scatter<<<wv(EA), 256, 0, stream>>>(f_x1, jA, iA, f_ea, out, EA, D600,
                                            EH + D600, EH);
  k_relu_region<<<ew(Nn * D600), 256, 0, stream>>>(out, Nn, D600, EH + D600, EH);
}
